// GCNClassifier_33217277067780
// MI455X (gfx1250) — compile-verified, matrix operand order checked by reading, not run on hardware
//
#include <hip/hip_runtime.h>

typedef __attribute__((ext_vector_type(16))) _Float16 v16h;
typedef __attribute__((ext_vector_type(8)))  float    v8f;

// ---- problem constants ------------------------------------------------------
constexpr int BB   = 64;    // batch
constexpr int LL   = 256;   // sequence length
constexpr int DIN  = 768;   // input dim
constexpr int MEMD = 300;   // GAT hidden dim
constexpr int NH   = 6;     // heads
constexpr int TOPK = 1024;  // top-k for adjacency
constexpr int NCLS = 3;     // classes
constexpr int NLAY = 2;     // GAT layers
constexpr int DKG  = 50;    // MEMD / NH (GAT head dim)
constexpr int DKM  = 128;   // DIN / NH (MHA head dim)
constexpr int BL   = BB * LL;

// ---- workspace layout (float elements) -------------------------------------
constexpr size_t SZ_QK    = (size_t)BL * DIN;        // 12,582,912
constexpr size_t OFF_Q    = 0;
constexpr size_t OFF_K    = SZ_QK;
constexpr size_t SZ_P     = (size_t)BB * LL * LL;    //  4,194,304
constexpr size_t OFF_P    = 2 * SZ_QK;               // p_attn
constexpr size_t OFF_ADJ  = OFF_P + SZ_P;
constexpr size_t OFF_KTH  = OFF_ADJ + SZ_P;          // 64 floats
// q/k are dead after the scores kernel -> overlay later buffers there:
constexpr size_t SZ_H     = (size_t)BL * MEMD;       //  4,915,200
constexpr size_t OFF_H    = 0;
constexpr size_t OFF_PROJ = OFF_H + SZ_H;
constexpr size_t OFF_HOUT = OFF_PROJ + SZ_H;
constexpr size_t OFF_SI   = OFF_HOUT + SZ_H;
constexpr size_t OFF_SJ   = OFF_SI + (size_t)NH * BL;

// ---- helpers ---------------------------------------------------------------
// K index of element e of an f16 A/B fragment (16x32 / 32x16), ISA 7.12.2
__device__ __forceinline__ int frag_k(int e, int hlf) {
  return (e < 8 ? e : e + 8) + hlf * 8;
}
__device__ __forceinline__ v8f zero_v8f() {
  v8f z;
#pragma unroll
  for (int i = 0; i < 8; ++i) z[i] = 0.f;
  return z;
}
// Vectorized fragment load from a row-major f32 row: elements 0..7 come from
// row[kk + 8*hlf .. +7], elements 8..15 from row[kk + 16 + 8*hlf .. +7].
// Both runs are contiguous -> 4x b128 loads. Requires 16B alignment of the
// runs (all row strides used here are multiples of 4 floats; kk % 32 == 0).
__device__ __forceinline__ v16h load_frag_row_f32(const float* __restrict__ row,
                                                  int kk, int hlf) {
  const float4* p0 = (const float4*)(row + kk + hlf * 8);
  const float4* p1 = (const float4*)(row + kk + 16 + hlf * 8);
  const float4 a0 = p0[0], a1 = p0[1], b0 = p1[0], b1 = p1[1];
  v16h f;
  f[0]  = (_Float16)a0.x; f[1]  = (_Float16)a0.y;
  f[2]  = (_Float16)a0.z; f[3]  = (_Float16)a0.w;
  f[4]  = (_Float16)a1.x; f[5]  = (_Float16)a1.y;
  f[6]  = (_Float16)a1.z; f[7]  = (_Float16)a1.w;
  f[8]  = (_Float16)b0.x; f[9]  = (_Float16)b0.y;
  f[10] = (_Float16)b0.z; f[11] = (_Float16)b0.w;
  f[12] = (_Float16)b1.x; f[13] = (_Float16)b1.y;
  f[14] = (_Float16)b1.z; f[15] = (_Float16)b1.w;
  return f;
}

// ---- generic WMMA GEMM: C[M,N] = A[M,K] @ W[K,N] + bias[N] ------------------
// Requires M % 64 == 0 (true for all call sites: M = 16384).
// block = 128 threads = 4 waves. Each wave computes a 64x32 tile:
// 4 m-tiles x 2 n-tiles -> 8 WMMAs per (4 vectorized A-frag + 2 B-frag) loads.
__global__ __launch_bounds__(128) void wmma_gemm_bias_kernel(
    const float* __restrict__ A, const float* __restrict__ W,
    const float* __restrict__ bias, float* __restrict__ C,
    int M, int N, int K)
{
  const int wave = threadIdx.x >> 5;
  const int lane = threadIdx.x & 31;
  const int hlf  = lane >> 4;
  const int lmod = lane & 15;
  const int mbase = blockIdx.x * 64;
  const int n0    = (blockIdx.y * 4 + wave) * 32;
  if (n0 >= N) return;   // wave-uniform; kernel has no LDS/barriers

  const int  nidx[2] = { n0 + lmod, n0 + 16 + lmod };
  const bool nok[2]  = { nidx[0] < N, nidx[1] < N };
  const int  ncl[2]  = { nok[0] ? nidx[0] : N - 1, nok[1] ? nidx[1] : N - 1 };

  v8f acc[4][2];
#pragma unroll
  for (int mt = 0; mt < 4; ++mt)
#pragma unroll
    for (int nt = 0; nt < 2; ++nt) acc[mt][nt] = zero_v8f();

  const int K32 = K & ~31;
  for (int kk = 0; kk < K32; kk += 32) {
    v16h bf[2];
#pragma unroll
    for (int nt = 0; nt < 2; ++nt) {
#pragma unroll
      for (int e = 0; e < 16; ++e) {
        const float wv = W[(size_t)(kk + frag_k(e, hlf)) * N + ncl[nt]];
        bf[nt][e] = (_Float16)(nok[nt] ? wv : 0.f);
      }
    }
#pragma unroll
    for (int mt = 0; mt < 4; ++mt) {
      const v16h af =
          load_frag_row_f32(A + (size_t)(mbase + mt * 16 + lmod) * K, kk, hlf);
#pragma unroll
      for (int nt = 0; nt < 2; ++nt)
        acc[mt][nt] = __builtin_amdgcn_wmma_f32_16x16x32_f16(
            false, af, false, bf[nt], (short)0, acc[mt][nt], false, false);
    }
  }
  if (K32 < K) {  // ragged K tail (K = 300 call sites)
    v16h bf[2];
#pragma unroll
    for (int nt = 0; nt < 2; ++nt) {
#pragma unroll
      for (int e = 0; e < 16; ++e) {
        const int kidx = K32 + frag_k(e, hlf);
        const int kcl  = (kidx < K) ? kidx : K - 1;
        const float wv = W[(size_t)kcl * N + ncl[nt]];
        bf[nt][e] = (_Float16)((kidx < K && nok[nt]) ? wv : 0.f);
      }
    }
#pragma unroll
    for (int mt = 0; mt < 4; ++mt) {
      const float* arow = A + (size_t)(mbase + mt * 16 + lmod) * K;
      v16h af;
#pragma unroll
      for (int e = 0; e < 16; ++e) {
        const int kidx = K32 + frag_k(e, hlf);
        const int kcl  = (kidx < K) ? kidx : K - 1;
        const float av = arow[kcl];
        af[e] = (_Float16)((kidx < K) ? av : 0.f);
      }
#pragma unroll
      for (int nt = 0; nt < 2; ++nt)
        acc[mt][nt] = __builtin_amdgcn_wmma_f32_16x16x32_f16(
            false, af, false, bf[nt], (short)0, acc[mt][nt], false, false);
    }
  }
  // epilogue: stores coalesced across lanes (consecutive n per lmod)
#pragma unroll
  for (int nt = 0; nt < 2; ++nt) {
    if (!nok[nt]) continue;
    const float bv = bias[nidx[nt]];
#pragma unroll
    for (int mt = 0; mt < 4; ++mt)
#pragma unroll
      for (int r = 0; r < 8; ++r)
        C[(size_t)(mbase + mt * 16 + r + hlf * 8) * N + nidx[nt]] =
            acc[mt][nt][r] + bv;
  }
}

// ---- fused scores: p_attn[b,l,m] = mean_h softmax_m(q_h . k_h / sqrt(128)) --
// one wave per (b, 16-row tile); loops heads; 16 column tiles kept in 16 f32
// accumulator fragments; softmax reduced across 16-lane halves with shfl_xor.
// All fragment loads are vectorized (head slice is contiguous, DKM=128).
__global__ __launch_bounds__(32) void scores_softmax_mean_kernel(
    const float* __restrict__ q, const float* __restrict__ k,
    float* __restrict__ p_attn)
{
  const int mtile = blockIdx.x;
  const int b     = blockIdx.y;
  const int lane  = threadIdx.x;
  const int hlf   = lane >> 4;
  const int lmod  = lane & 15;
  const float scale = 0.08838834764831845f;   // 1/sqrt(128)
  const int arow = mtile * 16 + lmod;

  for (int h = 0; h < NH; ++h) {
    const float* qb = q + (size_t)b * LL * DIN + (size_t)h * DKM;
    const float* kb = k + (size_t)b * LL * DIN + (size_t)h * DKM;
    v8f acc[16];
#pragma unroll
    for (int t = 0; t < 16; ++t) acc[t] = zero_v8f();

    for (int kk = 0; kk < DKM; kk += 32) {
      const v16h af = load_frag_row_f32(qb + (size_t)arow * DIN, kk, hlf);
#pragma unroll
      for (int t = 0; t < 16; ++t) {
        const int bcol = t * 16 + lmod;                  // B fragment: lane = n
        const v16h bf = load_frag_row_f32(kb + (size_t)bcol * DIN, kk, hlf);
        acc[t] = __builtin_amdgcn_wmma_f32_16x16x32_f16(
            false, af, false, bf, (short)0, acc[t], false, false);
      }
    }
    // row softmax: row of VGPR r in this half = mtile*16 + r + hlf*8;
    // its 256 columns live in 16 fragments x 16 lanes of this half.
#pragma unroll
    for (int r = 0; r < 8; ++r) {
      float mx = -3.4e38f;
#pragma unroll
      for (int t = 0; t < 16; ++t) mx = fmaxf(mx, acc[t][r]);
      for (int d = 1; d < 16; d <<= 1) mx = fmaxf(mx, __shfl_xor(mx, d, 32));
      float s = 0.f;
#pragma unroll
      for (int t = 0; t < 16; ++t) {
        const float e0 = __expf((acc[t][r] - mx) * scale);
        acc[t][r] = e0;
        s += e0;
      }
      for (int d = 1; d < 16; d <<= 1) s += __shfl_xor(s, d, 32);
      const float inv = 1.f / (s * (float)NH);           // fold head-mean in
#pragma unroll
      for (int t = 0; t < 16; ++t) acc[t][r] *= inv;
    }
    // this wave exclusively owns these 16 rows -> plain RMW accumulate
#pragma unroll
    for (int r = 0; r < 8; ++r) {
      const int lrow = mtile * 16 + r + hlf * 8;
#pragma unroll
      for (int t = 0; t < 16; ++t) {
        float* p = p_attn + ((size_t)b * LL + lrow) * LL + t * 16 + lmod;
        if (h == 0) *p = acc[t][r]; else *p += acc[t][r];
      }
    }
  }
}

// ---- exact k-th largest per batch via radix binary search on float bits -----
// values are softmax means >= 0 -> uint32 bit pattern is monotonic.
__global__ __launch_bounds__(256) void topk_threshold_kernel(
    const float* __restrict__ p_attn, float* __restrict__ kth)
{
  const int b = blockIdx.x, tid = threadIdx.x;
  const float* p = p_attn + (size_t)b * LL * LL;
  __shared__ int cnt[256];
  unsigned prefix = 0u;
  for (int bit = 31; bit >= 0; --bit) {
    const unsigned cand = prefix | (1u << bit);
    int c = 0;
    for (int i = tid; i < LL * LL; i += 256)
      c += (__float_as_uint(p[i]) >= cand) ? 1 : 0;
    cnt[tid] = c;
    __syncthreads();
    for (int s = 128; s > 0; s >>= 1) {
      if (tid < s) cnt[tid] += cnt[tid + s];
      __syncthreads();
    }
    const int total = cnt[0];
    __syncthreads();
    if (total >= TOPK) prefix = cand;
  }
  if (tid == 0) kth[b] = __uint_as_float(prefix);
}

// adj nonzero iff i==j or p[i,j]>=kth or p[j,i]>=kth  (a + a^T, diag=1)
__global__ __launch_bounds__(256) void build_adj_kernel(
    const float* __restrict__ p_attn, const float* __restrict__ kth,
    float* __restrict__ adj)
{
  const int idx = blockIdx.x * 256 + threadIdx.x;     // < BB*LL*LL = 4M
  const int b = idx / (LL * LL);
  const int rem = idx % (LL * LL);
  const int i = rem / LL, j = rem % LL;
  const float t = kth[b];
  const float* p = p_attn + (size_t)b * LL * LL;
  adj[(size_t)idx] = (i == j || p[i * LL + j] >= t || p[j * LL + i] >= t) ? 1.f : 0.f;
}

// s_i = proj @ aw[:DK], s_j = proj @ aw[DK:]  over (NH, B*L) rows
__global__ __launch_bounds__(256) void si_sj_kernel(
    const float* __restrict__ proj, const float* __restrict__ aw_layer,
    float* __restrict__ s_i, float* __restrict__ s_j)
{
  const int idx = blockIdx.x * 256 + threadIdx.x;     // < NH*BL
  const float* p = proj + (size_t)idx * DKG;
  float a = 0.f, c = 0.f;
  for (int kq = 0; kq < DKG; ++kq) {
    a += p[kq] * aw_layer[kq];
    c += p[kq] * aw_layer[DKG + kq];
  }
  s_i[idx] = a;
  s_j[idx] = c;
}

// ---- fused GAT attention + head_out = softmax(mask(leaky(si+sj+ab))) @ proj -
// one wave per (h, b, 16-row tile). Pass 1: masked row max & sum (lane pairs
// split the 256 columns, combined with shfl_xor 16). Pass 2: attention weights
// recomputed into f16 A-fragments, proj gathered into B-fragments, 4 WMMA
// n-tiles cover DK=50 (padded to 64).
__global__ __launch_bounds__(32) void gat_attn_head_out_kernel(
    const float* __restrict__ proj, const float* __restrict__ s_i,
    const float* __restrict__ s_j, const float* __restrict__ adj,
    const float* __restrict__ ab_all, int layer,
    float* __restrict__ head_out)
{
  const int mtile = blockIdx.x;
  const int b = blockIdx.y;
  const int h = blockIdx.z;
  const int lane = threadIdx.x;
  const int hlf  = lane >> 4;
  const int lmod = lane & 15;
  const float abv = ab_all[layer];

  const float* sI    = s_i + (size_t)h * BL + (size_t)b * LL;
  const float* sJ    = s_j + (size_t)h * BL + (size_t)b * LL;
  const float* adjb  = adj + (size_t)b * LL * LL;
  const float* projb = proj + ((size_t)h * BL + (size_t)b * LL) * DKG;

  const int row = mtile * 16 + lmod;
  const float si = sI[row];

  // pass 1: row max / sumexp (masked entries contribute exactly 0, diag open)
  float mx = -3.4e38f;
  for (int m = hlf * 128; m < hlf * 128 + 128; ++m) {
    if (adjb[row * LL + m] != 0.f) {
      float sc = si + sJ[m] + abv;
      sc = (sc > 0.f) ? sc : 0.2f * sc;
      mx = fmaxf(mx, sc);
    }
  }
  mx = fmaxf(mx, __shfl_xor(mx, 16, 32));
  float sum = 0.f;
  for (int m = hlf * 128; m < hlf * 128 + 128; ++m) {
    if (adjb[row * LL + m] != 0.f) {
      float sc = si + sJ[m] + abv;
      sc = (sc > 0.f) ? sc : 0.2f * sc;
      sum += __expf(sc - mx);
    }
  }
  sum += __shfl_xor(sum, 16, 32);
  const float inv = 1.f / sum;

  // pass 2: D(16 x 64) = attn(16 x 256) @ proj(256 x 50-padded)
  v8f acc[4];
#pragma unroll
  for (int nt = 0; nt < 4; ++nt) acc[nt] = zero_v8f();

  for (int ms = 0; ms < LL; ms += 32) {
    v16h af;
#pragma unroll
    for (int e = 0; e < 16; ++e) {
      const int m = ms + frag_k(e, hlf);
      float a0 = 0.f;
      if (adjb[row * LL + m] != 0.f) {
        float sc = si + sJ[m] + abv;
        sc = (sc > 0.f) ? sc : 0.2f * sc;
        a0 = __expf(sc - mx) * inv;
      }
      af[e] = (_Float16)a0;
    }
#pragma unroll
    for (int nt = 0; nt < 4; ++nt) {
      const int kcol = nt * 16 + lmod;
      const int kc = (kcol < DKG) ? kcol : 0;
      v16h bf;
#pragma unroll
      for (int e = 0; e < 16; ++e) {
        const int m = ms + frag_k(e, hlf);
        const float v = projb[(size_t)m * DKG + kc];
        bf[e] = (_Float16)((kcol < DKG) ? v : 0.f);
      }
      acc[nt] = __builtin_amdgcn_wmma_f32_16x16x32_f16(
          false, af, false, bf, (short)0, acc[nt], false, false);
    }
  }
#pragma unroll
  for (int nt = 0; nt < 4; ++nt) {
    const int kcol = nt * 16 + lmod;
    if (kcol < DKG) {
#pragma unroll
      for (int r = 0; r < 8; ++r) {
        const int lrow = mtile * 16 + r + hlf * 8;
        head_out[((size_t)h * BL + (size_t)b * LL + lrow) * DKG + kcol] = acc[nt][r];
      }
    }
  }
}

// ---- residual + LayerNorm over MEM=300, in-place on h ----------------------
__global__ __launch_bounds__(256) void residual_ln_kernel(
    float* __restrict__ hbuf, const float* __restrict__ head_out,
    const float* __restrict__ g_ln, const float* __restrict__ b_ln)
{
  const int row = blockIdx.x;     // b*L + l
  const int tid = threadIdx.x;
  __shared__ float rbuf[MEMD];
  __shared__ float ssum[256], ssq[256];
  float lsum = 0.f, lsq = 0.f;
  for (int i = tid; i < MEMD; i += 256) {
    const int hh = i / DKG, kq = i % DKG;
    const float cat = head_out[((size_t)hh * BL + row) * DKG + kq];
    const float r = hbuf[(size_t)row * MEMD + i] + cat;
    rbuf[i] = r;
    lsum += r;
    lsq += r * r;
  }
  ssum[tid] = lsum;
  ssq[tid] = lsq;
  __syncthreads();
  for (int s = 128; s > 0; s >>= 1) {
    if (tid < s) { ssum[tid] += ssum[tid + s]; ssq[tid] += ssq[tid + s]; }
    __syncthreads();
  }
  const float mu  = ssum[0] / (float)MEMD;
  const float var = ssq[0] / (float)MEMD - mu * mu;
  const float rstd = rsqrtf(var + 1e-5f);
  for (int i = tid; i < MEMD; i += 256)
    hbuf[(size_t)row * MEMD + i] = (rbuf[i] - mu) * rstd * g_ln[i] + b_ln[i];
}

// ---- mean pool over L, then (300 x 3) classifier ---------------------------
__global__ __launch_bounds__(256) void pool_classify_kernel(
    const float* __restrict__ hbuf, const float* __restrict__ Wc,
    const float* __restrict__ bc, float* __restrict__ out)
{
  const int b = blockIdx.x;
  const int tid = threadIdx.x;
  __shared__ float pooled[MEMD];
  for (int i = tid; i < MEMD; i += 256) {
    float s = 0.f;
    for (int l = 0; l < LL; ++l) s += hbuf[((size_t)b * LL + l) * MEMD + i];
    pooled[i] = s / (float)LL;
  }
  __syncthreads();
  if (tid < NCLS) {
    float s = bc[tid];
    for (int i = 0; i < MEMD; ++i) s += pooled[i] * Wc[i * NCLS + tid];
    out[b * NCLS + tid] = s;
  }
}

// ---- host orchestration ----------------------------------------------------
extern "C" void kernel_launch(void* const* d_in, const int* in_sizes, int n_in,
                              void* d_out, int out_size, void* d_ws, size_t ws_size,
                              hipStream_t stream) {
  (void)in_sizes; (void)n_in; (void)out_size; (void)ws_size;
  const float* x    = (const float*)d_in[0];
  const float* wq   = (const float*)d_in[1];
  const float* bq   = (const float*)d_in[2];
  const float* wk   = (const float*)d_in[3];
  const float* bk   = (const float*)d_in[4];
  const float* w_in = (const float*)d_in[5];
  const float* b_in = (const float*)d_in[6];
  const float* Wh   = (const float*)d_in[7];
  const float* bh   = (const float*)d_in[8];
  const float* aw   = (const float*)d_in[9];
  const float* ab   = (const float*)d_in[10];
  const float* g_ln = (const float*)d_in[11];
  const float* b_ln = (const float*)d_in[12];
  const float* Wc   = (const float*)d_in[13];
  const float* bc   = (const float*)d_in[14];

  float* ws       = (float*)d_ws;
  float* q        = ws + OFF_Q;
  float* kmat     = ws + OFF_K;
  float* p_attn   = ws + OFF_P;
  float* adj      = ws + OFF_ADJ;
  float* kth      = ws + OFF_KTH;
  float* hbuf     = ws + OFF_H;     // overlays q (dead after scores)
  float* proj     = ws + OFF_PROJ;
  float* head_out = ws + OFF_HOUT;
  float* s_i      = ws + OFF_SI;
  float* s_j      = ws + OFF_SJ;

  // gemm grid: 64 rows per block.x, 4 waves * 32 cols per block.y
  auto gemm_grid = [](int M, int N) {
    return dim3((unsigned)(M / 64), (unsigned)(((N + 31) / 32 + 3) / 4));
  };

  // 1) q = x@wq+bq, k = x@wk+bk   (M=16384, K=768, N=768)
  wmma_gemm_bias_kernel<<<gemm_grid(BL, DIN), 128, 0, stream>>>(
      x, wq, bq, q, BL, DIN, DIN);
  wmma_gemm_bias_kernel<<<gemm_grid(BL, DIN), 128, 0, stream>>>(
      x, wk, bk, kmat, BL, DIN, DIN);

  // 2) p_attn = mean_h softmax(q k^T / sqrt(128))  (fused, no score tensor)
  scores_softmax_mean_kernel<<<dim3(LL / 16, BB), 32, 0, stream>>>(q, kmat, p_attn);

  // 3) exact top-k threshold per batch, then symmetric+diag adjacency
  topk_threshold_kernel<<<BB, 256, 0, stream>>>(p_attn, kth);
  build_adj_kernel<<<(BB * LL * LL) / 256, 256, 0, stream>>>(p_attn, kth, adj);

  // 4) h = x @ w_in + b_in   (M=16384, K=768, N=300)
  wmma_gemm_bias_kernel<<<gemm_grid(BL, MEMD), 128, 0, stream>>>(
      x, w_in, b_in, hbuf, BL, MEMD, DIN);

  // 5) GAT layers
  for (int layer = 0; layer < NLAY; ++layer) {
    for (int hh = 0; hh < NH; ++hh) {
      wmma_gemm_bias_kernel<<<gemm_grid(BL, DKG), 128, 0, stream>>>(
          hbuf,
          Wh + (size_t)(layer * NH + hh) * MEMD * DKG,
          bh + (size_t)(layer * NH + hh) * DKG,
          proj + (size_t)hh * BL * DKG,
          BL, DKG, MEMD);
    }
    si_sj_kernel<<<(NH * BL) / 256, 256, 0, stream>>>(
        proj, aw + (size_t)layer * 2 * DKG, s_i, s_j);
    gat_attn_head_out_kernel<<<dim3(LL / 16, BB, NH), 32, 0, stream>>>(
        proj, s_i, s_j, adj, ab, layer, head_out);
    residual_ln_kernel<<<BL, 256, 0, stream>>>(
        hbuf, head_out, g_ln + (size_t)layer * MEMD, b_ln + (size_t)layer * MEMD);
  }

  // 6) pooled @ Wc + bc  ->  (64, 3) output
  pool_classify_kernel<<<BB, 256, 0, stream>>>(hbuf, Wc, bc, (float*)d_out);
}
